// SparseAttention_81724637708864
// MI455X (gfx1250) — matrix operand order repaired and checked by reading, not compiled, not run on hardware
//
#include <hip/hip_runtime.h>
#include <hip/hip_bf16.h>
#include <stdint.h>

#define DM   1024
#define NH   16
#define HD   64
#define BATCH 2
#define SEQ  2048
#define MTOT (BATCH*SEQ)   // 4096 rows for all GEMMs

typedef __attribute__((ext_vector_type(16))) __bf16        v16bf;
typedef __attribute__((ext_vector_type(8)))  float         v8f;
typedef __attribute__((ext_vector_type(8)))  unsigned short v8us;
typedef __attribute__((ext_vector_type(4)))  int           v4i;

union Frag {
    v16bf bf;
    struct { v8us lo; v8us hi; } h;
};

// ---- CDNA5 async global->LDS copy (confirmed signature: int4 AS1*, int4 AS3*, imm, imm)
#if defined(__gfx1250__) && __has_builtin(__builtin_amdgcn_global_load_async_to_lds_b128)
#define ASYNC_LDS 1
#define GPTR(p) ((__attribute__((address_space(1))) v4i*)(p))
#define LPTR(p) ((__attribute__((address_space(3))) v4i*)(p))
#else
#define ASYNC_LDS 0
#endif

__device__ __forceinline__ void copy16(const unsigned short* g, unsigned short* l) {
#if ASYNC_LDS
    __builtin_amdgcn_global_load_async_to_lds_b128(GPTR(g), LPTR(l), 0, 0);
#else
    *(v8us*)l = *(const v8us*)g;
#endif
}
__device__ __forceinline__ void copy_wait() {
#if ASYNC_LDS
    asm volatile("s_wait_asynccnt 0" ::: "memory");
#endif
}

// ---- packed f32 -> bf16 (v_cvt_pk_bf16_f32 when available, manual RNE fallback)
#if defined(__gfx1250__) && __has_builtin(__builtin_amdgcn_cvt_pk_bf16_f32)
#define HAVE_PK_BF16 1
typedef __attribute__((ext_vector_type(2))) __bf16 v2bf;
#else
#define HAVE_PK_BF16 0
#endif

__device__ __forceinline__ unsigned int pk2bfu(float lo, float hi) {
#if HAVE_PK_BF16
    v2bf t = __builtin_amdgcn_cvt_pk_bf16_f32(lo, hi);
    return __builtin_bit_cast(unsigned int, t);
#else
    union { float f; uint32_t u; } a, b; a.f = lo; b.f = hi;
    uint32_t ra = a.u + 0x7FFFu + ((a.u >> 16) & 1u);
    uint32_t rb = b.u + 0x7FFFu + ((b.u >> 16) & 1u);
    return (ra >> 16) | (rb & 0xFFFF0000u);
#endif
}
__device__ __forceinline__ unsigned short f2bfu(float f) {
    return (unsigned short)(pk2bfu(f, f) & 0xFFFFu);
}

// ---------------------------------------------------------------------------
// f32 -> bf16 bulk conversion, 2 elements/thread (one v_cvt_pk per thread)
// ---------------------------------------------------------------------------
__global__ void cvt_bf16x2_kernel(const float* __restrict__ src,
                                  unsigned int* __restrict__ dst, int npairs) {
    int i = blockIdx.x * blockDim.x + threadIdx.x;
    if (i < npairs) {
        const float2 v = ((const float2*)src)[i];
        dst[i] = pk2bfu(v.x, v.y);
    }
}

// ---------------------------------------------------------------------------
// Pack boolean mask (1 byte/elem) into 32-bit words via wave32 ballot.
// ---------------------------------------------------------------------------
__global__ void pack_mask_kernel(const unsigned char* __restrict__ mask,
                                 unsigned int* __restrict__ mw, int nwords) {
    int lane = threadIdx.x & 31;
    int wave = blockIdx.x * (blockDim.x >> 5) + (threadIdx.x >> 5);
    if (wave >= nwords) return;
    unsigned char mv = mask[(size_t)wave * 32 + lane];
    unsigned long long bal = __ballot(mv != 0);
    if (lane == 0) mw[wave] = (unsigned int)bal;
}

// ---------------------------------------------------------------------------
// GEMM:  Y(4096, 1024) = Xbf(bf16) @ W^T(bf16) + bias
// mode 0: bf16 per-head [b,h,s,d]            (Q/K projections)
// mode 2: bf16 per-head TRANSPOSED [b,h,d,s] (V projection)
// mode 1: f32 flat (M,1024) -> d_out         (output projection)
// Double-buffered LDS staging via async global->LDS copies.
// ---------------------------------------------------------------------------
__global__ __launch_bounds__(256) void proj_gemm(
    const unsigned short* __restrict__ Xbf,
    const unsigned short* __restrict__ Wbf,
    const float* __restrict__ bias,
    unsigned short* __restrict__ out_bf,
    float* __restrict__ out_f32,
    int mode) {

    __shared__ __align__(16) unsigned short Xs[2][64][32];
    __shared__ __align__(16) unsigned short Ws[2][64][32];

    const int tid   = threadIdx.x;
    const int lane  = tid & 31;
    const int half  = lane >> 4;
    const int l16   = lane & 15;
    const int wave  = tid >> 5;
    const int m_off = (wave >> 1) * 16;
    const int n_off = (wave & 1) * 32;
    const int m_base = blockIdx.y * 64;
    const int n_base = blockIdx.x * 64;

    const int ldr = tid >> 2;               // 0..63
    const int ldc = (tid & 3) * 8;          // 0,8,16,24

    const unsigned short* xrow = Xbf + (size_t)(m_base + ldr) * DM + ldc;
    const unsigned short* wrow = Wbf + (size_t)(n_base + ldr) * DM + ldc;

    auto issue_tile = [&](int k0, int buf) {
        copy16(xrow + k0, &Xs[buf][ldr][ldc]);
        copy16(wrow + k0, &Ws[buf][ldr][ldc]);
    };

    v8f c0 = {0,0,0,0,0,0,0,0};
    v8f c1 = {0,0,0,0,0,0,0,0};

    issue_tile(0, 0);
    copy_wait();
    __syncthreads();

    const int NKT = DM / 32;               // 32 K-tiles
    for (int kt = 0; kt < NKT; ++kt) {
        const int cur = kt & 1;
        if (kt + 1 < NKT) issue_tile((kt + 1) * 32, cur ^ 1);

        Frag a, b0, b1;
        a.h.lo  = *(const v8us*)&Xs[cur][m_off + l16][8 * half];
        a.h.hi  = *(const v8us*)&Xs[cur][m_off + l16][16 + 8 * half];
        b0.h.lo = *(const v8us*)&Ws[cur][n_off + l16][8 * half];
        b0.h.hi = *(const v8us*)&Ws[cur][n_off + l16][16 + 8 * half];
        b1.h.lo = *(const v8us*)&Ws[cur][n_off + 16 + l16][8 * half];
        b1.h.hi = *(const v8us*)&Ws[cur][n_off + 16 + l16][16 + 8 * half];

        c0 = __builtin_amdgcn_wmma_f32_16x16x32_bf16(false, a.bf, false, b0.bf,
                                                     (short)0, c0, false, false);
        c1 = __builtin_amdgcn_wmma_f32_16x16x32_bf16(false, a.bf, false, b1.bf,
                                                     (short)0, c1, false, false);

        if (kt + 1 < NKT) {
            copy_wait();
            __syncthreads();
        }
    }

    const int n0 = n_base + n_off + l16;
    const int n1 = n0 + 16;
    const float bv0 = bias[n0];
    const float bv1 = bias[n1];
#pragma unroll
    for (int j = 0; j < 8; ++j) {
        const int m = m_base + m_off + j + 8 * half;
        const float y0 = c0[j] + bv0;
        const float y1 = c1[j] + bv1;
        const int bb = m >> 11;
        const int s  = m & (SEQ - 1);
        if (mode == 0) {
            const int h0 = n0 >> 6, d0 = n0 & 63;
            const int h1 = n1 >> 6, d1 = n1 & 63;
            out_bf[(((size_t)bb * NH + h0) * SEQ + s) * HD + d0] = f2bfu(y0);
            out_bf[(((size_t)bb * NH + h1) * SEQ + s) * HD + d1] = f2bfu(y1);
        } else if (mode == 2) {   // transposed per-head: [b,h,d,s]
            const int h0 = n0 >> 6, d0 = n0 & 63;
            const int h1 = n1 >> 6, d1 = n1 & 63;
            out_bf[(((size_t)bb * NH + h0) * HD + d0) * SEQ + s] = f2bfu(y0);
            out_bf[(((size_t)bb * NH + h1) * HD + d1) * SEQ + s] = f2bfu(y1);
        } else {
            out_f32[(size_t)m * DM + n0] = y0;
            out_f32[(size_t)m * DM + n1] = y1;
        }
    }
}

// ---------------------------------------------------------------------------
// Flash attention: block = 4 waves = 64 q-rows of one (b,h).
// kv tiles of 64: 8 WMMAs scores + 8 WMMAs P*V per tile.
// Double-buffered async K/V staging; output written in bf16 for the O-proj.
// ---------------------------------------------------------------------------
__global__ __launch_bounds__(128) void attn_kernel(
    const unsigned short* __restrict__ Qb,    // (B,H,S,hd) bf16
    const unsigned short* __restrict__ Kb,    // (B,H,S,hd) bf16
    const unsigned short* __restrict__ VbT,   // (B,H,hd,S) bf16  (pre-transposed)
    const unsigned int*   __restrict__ maskw, // (B,S,S/32) bit-packed
    unsigned short* __restrict__ outbf) {     // (B,S,D) bf16

    __shared__ __align__(16) unsigned short Ks[2][64][64];   // [kv][d]
    __shared__ __align__(16) unsigned short Vt[2][64][64];   // [d][kv]
    __shared__ __align__(16) unsigned short Ps[4][16][64];

    const int tid  = threadIdx.x;
    const int lane = tid & 31;
    const int half = lane >> 4;
    const int l16  = lane & 15;
    const int wave = tid >> 5;

    const int bh = blockIdx.y;
    const int b  = bh >> 4;
    const int h  = bh & 15;
    const int qrow0 = blockIdx.x * 64 + wave * 16;

    // Q fragments (16x64) live in registers for the whole kv loop
    const unsigned short* qrow = Qb + ((size_t)bh * SEQ + qrow0 + l16) * HD;
    Frag qa0, qa1;
    qa0.h.lo = *(const v8us*)(qrow + 8 * half);
    qa0.h.hi = *(const v8us*)(qrow + 16 + 8 * half);
    qa1.h.lo = *(const v8us*)(qrow + 32 + 8 * half);
    qa1.h.hi = *(const v8us*)(qrow + 48 + 8 * half);

    float m_r[8], l_r[8];
#pragma unroll
    for (int j = 0; j < 8; ++j) { m_r[j] = -3.0e38f; l_r[j] = 0.0f; }
    v8f o0 = {0,0,0,0,0,0,0,0}, o1 = o0, o2 = o0, o3 = o0;

    const int ldr = tid >> 1;            // 0..63
    const int ldc = (tid & 1) * 32;      // 0 or 32

    auto issue_tile = [&](int kt_, int buf) {
        const int kvb = kt_ * 64;
        const unsigned short* ksrc = Kb  + ((size_t)bh * SEQ + kvb + ldr) * HD + ldc;
        const unsigned short* vsrc = VbT + ((size_t)bh * HD + ldr) * SEQ + kvb + ldc;
#pragma unroll
        for (int i = 0; i < 4; ++i) {
            copy16(ksrc + 8 * i, &Ks[buf][ldr][ldc + 8 * i]);
            copy16(vsrc + 8 * i, &Vt[buf][ldr][ldc + 8 * i]);
        }
    };

    issue_tile(0, 0);
    copy_wait();
    __syncthreads();

    const int NT = SEQ / 64;             // 32 kv tiles
    for (int kt = 0; kt < NT; ++kt) {
        const int cur = kt & 1;
        if (kt + 1 < NT) issue_tile(kt + 1, cur ^ 1);

        // --- scores: s_n covers kv columns n*16..n*16+15 ---
        v8f s0 = {0,0,0,0,0,0,0,0}, s1 = s0, s2 = s0, s3 = s0;
        Frag kb;
#define SCORE(SACC, NT_)                                                       \
        kb.h.lo = *(const v8us*)&Ks[cur][(NT_)*16 + l16][8 * half];            \
        kb.h.hi = *(const v8us*)&Ks[cur][(NT_)*16 + l16][16 + 8 * half];       \
        SACC = __builtin_amdgcn_wmma_f32_16x16x32_bf16(false, qa0.bf, false,   \
                   kb.bf, (short)0, SACC, false, false);                       \
        kb.h.lo = *(const v8us*)&Ks[cur][(NT_)*16 + l16][32 + 8 * half];       \
        kb.h.hi = *(const v8us*)&Ks[cur][(NT_)*16 + l16][48 + 8 * half];       \
        SACC = __builtin_amdgcn_wmma_f32_16x16x32_bf16(false, qa1.bf, false,   \
                   kb.bf, (short)0, SACC, false, false);
        SCORE(s0, 0) SCORE(s1, 1) SCORE(s2, 2) SCORE(s3, 3)
#undef SCORE

        // --- mask + online softmax; accumulator elem j = one q row ---
#pragma unroll
        for (int j = 0; j < 8; ++j) {
            const int q = qrow0 + j + 8 * half;
            const unsigned long long wp = *(const unsigned long long*)
                (maskw + ((size_t)b * SEQ + q) * (SEQ / 32) + 2 * kt);
            const unsigned int w0 = (unsigned int)wp;
            const unsigned int w1 = (unsigned int)(wp >> 32);
            const bool m0 = (w0 >> l16) & 1u;
            const bool m1 = (w0 >> (16 + l16)) & 1u;
            const bool m2 = (w1 >> l16) & 1u;
            const bool m3 = (w1 >> (16 + l16)) & 1u;
            const float a0 = m0 ? -1.0e30f : s0[j] * 0.125f;  // 1/sqrt(64)
            const float a1 = m1 ? -1.0e30f : s1[j] * 0.125f;
            const float a2 = m2 ? -1.0e30f : s2[j] * 0.125f;
            const float a3 = m3 ? -1.0e30f : s3[j] * 0.125f;

            float t = fmaxf(fmaxf(a0, a1), fmaxf(a2, a3));
#pragma unroll
            for (int d = 1; d < 16; d <<= 1) t = fmaxf(t, __shfl_xor(t, d, 16));
            const float mnew  = fmaxf(m_r[j], t);
            const float scale = __expf(m_r[j] - mnew);
            const float p0 = m0 ? 0.0f : __expf(a0 - mnew);
            const float p1 = m1 ? 0.0f : __expf(a1 - mnew);
            const float p2 = m2 ? 0.0f : __expf(a2 - mnew);
            const float p3 = m3 ? 0.0f : __expf(a3 - mnew);
            float ssum = (p0 + p1) + (p2 + p3);
#pragma unroll
            for (int d = 1; d < 16; d <<= 1) ssum += __shfl_xor(ssum, d, 16);
            l_r[j] = l_r[j] * scale + ssum;
            m_r[j] = mnew;
            o0[j] *= scale; o1[j] *= scale; o2[j] *= scale; o3[j] *= scale;

            const int pr = j + 8 * half;
            const unsigned int u01 = pk2bfu(p0, p1);
            const unsigned int u23 = pk2bfu(p2, p3);
            Ps[wave][pr][l16]      = (unsigned short)(u01 & 0xFFFFu);
            Ps[wave][pr][16 + l16] = (unsigned short)(u01 >> 16);
            Ps[wave][pr][32 + l16] = (unsigned short)(u23 & 0xFFFFu);
            Ps[wave][pr][48 + l16] = (unsigned short)(u23 >> 16);
        }

        // same-wave LDS RAW: P stores -> P fragment loads
        asm volatile("s_wait_dscnt 0" ::: "memory");

        // --- O += P(16x64) * V(64x64) ---
        Frag pa0, pa1, vb;
        pa0.h.lo = *(const v8us*)&Ps[wave][l16][8 * half];
        pa0.h.hi = *(const v8us*)&Ps[wave][l16][16 + 8 * half];
        pa1.h.lo = *(const v8us*)&Ps[wave][l16][32 + 8 * half];
        pa1.h.hi = *(const v8us*)&Ps[wave][l16][48 + 8 * half];
#define PV(OACC, DT)                                                           \
        vb.h.lo = *(const v8us*)&Vt[cur][(DT)*16 + l16][8 * half];             \
        vb.h.hi = *(const v8us*)&Vt[cur][(DT)*16 + l16][16 + 8 * half];        \
        OACC = __builtin_amdgcn_wmma_f32_16x16x32_bf16(false, pa0.bf, false,   \
                   vb.bf, (short)0, OACC, false, false);                       \
        vb.h.lo = *(const v8us*)&Vt[cur][(DT)*16 + l16][32 + 8 * half];        \
        vb.h.hi = *(const v8us*)&Vt[cur][(DT)*16 + l16][48 + 8 * half];        \
        OACC = __builtin_amdgcn_wmma_f32_16x16x32_bf16(false, pa1.bf, false,   \
                   vb.bf, (short)0, OACC, false, false);
        PV(o0, 0) PV(o1, 1) PV(o2, 2) PV(o3, 3)
#undef PV

        if (kt + 1 < NT) {
            copy_wait();
            __syncthreads();
        }
    }

    // --- epilogue: normalize + write (B,S,D) bf16 for the O-projection ---
#pragma unroll
    for (int j = 0; j < 8; ++j) {
        const float rl  = l_r[j];
        const float inv = rl > 0.0f ? 1.0f / rl : 0.0f;
        const int q = qrow0 + j + 8 * half;
        unsigned short* orow = outbf + ((size_t)b * SEQ + q) * DM + h * HD;
        orow[0  + l16] = f2bfu(o0[j] * inv);
        orow[16 + l16] = f2bfu(o1[j] * inv);
        orow[32 + l16] = f2bfu(o2[j] * inv);
        orow[48 + l16] = f2bfu(o3[j] * inv);
    }
}

// ---------------------------------------------------------------------------
extern "C" void kernel_launch(void* const* d_in, const int* in_sizes, int n_in,
                              void* d_out, int out_size, void* d_ws, size_t ws_size,
                              hipStream_t stream) {
    const float* query = (const float*)d_in[0];
    const float* key_t = (const float*)d_in[1];
    const float* value = (const float*)d_in[2];
    const unsigned char* mask = (const unsigned char*)d_in[3];
    const float* Wq = (const float*)d_in[4];
    const float* Wk = (const float*)d_in[5];
    const float* Wv = (const float*)d_in[6];
    const float* Wo = (const float*)d_in[7];
    const float* bq = (const float*)d_in[8];
    const float* bk = (const float*)d_in[9];
    const float* bv = (const float*)d_in[10];
    const float* bo = (const float*)d_in[11];
    float* out = (float*)d_out;

    char* ws = (char*)d_ws;
    size_t off = 0;
    auto carve = [&](size_t bytes) {
        char* p = ws + off;
        off += (bytes + 255) & ~(size_t)255;
        return p;
    };
    unsigned short* Wq_bf = (unsigned short*)carve((size_t)DM * DM * 2);
    unsigned short* Wk_bf = (unsigned short*)carve((size_t)DM * DM * 2);
    unsigned short* Wv_bf = (unsigned short*)carve((size_t)DM * DM * 2);
    unsigned short* Wo_bf = (unsigned short*)carve((size_t)DM * DM * 2);
    unsigned short* Xq_bf = (unsigned short*)carve((size_t)MTOT * DM * 2);
    unsigned short* Xk_bf = (unsigned short*)carve((size_t)MTOT * DM * 2);
    unsigned short* Xv_bf = (unsigned short*)carve((size_t)MTOT * DM * 2);
    unsigned short* Qb    = (unsigned short*)carve((size_t)MTOT * DM * 2);
    unsigned short* Kb    = (unsigned short*)carve((size_t)MTOT * DM * 2);
    unsigned short* VbT   = (unsigned short*)carve((size_t)MTOT * DM * 2);
    unsigned int*   mw    = (unsigned int*)carve((size_t)BATCH * SEQ * (SEQ / 32) * 4);
    unsigned short* attnb = (unsigned short*)carve((size_t)MTOT * DM * 2);

    // one-shot f32 -> bf16 conversions (weights + activations)
    const int wPairs = DM * DM / 2;
    cvt_bf16x2_kernel<<<(wPairs + 255) / 256, 256, 0, stream>>>(Wq, (unsigned int*)Wq_bf, wPairs);
    cvt_bf16x2_kernel<<<(wPairs + 255) / 256, 256, 0, stream>>>(Wk, (unsigned int*)Wk_bf, wPairs);
    cvt_bf16x2_kernel<<<(wPairs + 255) / 256, 256, 0, stream>>>(Wv, (unsigned int*)Wv_bf, wPairs);
    cvt_bf16x2_kernel<<<(wPairs + 255) / 256, 256, 0, stream>>>(Wo, (unsigned int*)Wo_bf, wPairs);
    const int xPairs = MTOT * DM / 2;
    cvt_bf16x2_kernel<<<(xPairs + 255) / 256, 256, 0, stream>>>(query, (unsigned int*)Xq_bf, xPairs);
    cvt_bf16x2_kernel<<<(xPairs + 255) / 256, 256, 0, stream>>>(key_t, (unsigned int*)Xk_bf, xPairs);
    cvt_bf16x2_kernel<<<(xPairs + 255) / 256, 256, 0, stream>>>(value, (unsigned int*)Xv_bf, xPairs);

    const int nwords = BATCH * SEQ * (SEQ / 32);       // 262144
    pack_mask_kernel<<<nwords / 8, 256, 0, stream>>>(mask, mw, nwords);

    dim3 ggrid(DM / 64, MTOT / 64);
    proj_gemm<<<ggrid, 256, 0, stream>>>(Xq_bf, Wq_bf, bq, Qb, nullptr, 0);
    proj_gemm<<<ggrid, 256, 0, stream>>>(Xk_bf, Wk_bf, bk, Kb, nullptr, 0);
    proj_gemm<<<ggrid, 256, 0, stream>>>(Xv_bf, Wv_bf, bv, VbT, nullptr, 2);

    attn_kernel<<<dim3(SEQ / 64, BATCH * NH), 128, 0, stream>>>(Qb, Kb, VbT, mw, attnb);

    proj_gemm<<<ggrid, 256, 0, stream>>>(attnb, Wo_bf, bo, nullptr, out, 1);
}
